// TransformerConv_24979529793770
// MI455X (gfx1250) — matrix operand order repaired and checked by reading, not compiled, not run on hardware
//
#include <hip/hip_runtime.h>
#include <hip/hip_bf16.h>

// ---------------------------------------------------------------------------
// MI455X (gfx1250) implementation of the masked-double-softmax attention block.
// All matmuls on v_wmma_f32_16x16x32_bf16 (wave32). Attention fused flash-style
// so the 268MB attn matrix never touches HBM. WMMA loops are explicitly
// double-buffered so k+1 fragment loads are in flight behind k's WMMAs.
// ---------------------------------------------------------------------------

typedef __bf16 bf16_t;
typedef __attribute__((ext_vector_type(16))) __bf16 v16bf;
typedef __attribute__((ext_vector_type(8)))  __bf16 bf16x8;
typedef __attribute__((ext_vector_type(4)))  __bf16 bf16x4;
typedef __attribute__((ext_vector_type(8)))  float  v8f;
typedef __attribute__((ext_vector_type(4)))  float  f32x4;
typedef __attribute__((ext_vector_type(4)))  int    i32x4;

union AB16 { v16bf v; bf16x8 h[2]; };

#define SLEN  512
#define BSZ   16
#define EMB   1024
#define NHEAD 16
#define HD    64
#define NEG1F (-9.0e15f)
#define NEG2F (-9.0e16f)

__device__ inline v8f wmma_bf16(v16bf a, v16bf b, v8f c) {
  return __builtin_amdgcn_wmma_f32_16x16x32_bf16(false, a, false, b,
                                                 (short)0, c, false, false);
}

// -------------------------------- convert ---------------------------------
__global__ __launch_bounds__(256) void f32_to_bf16_kernel(
    const float* __restrict__ src, bf16_t* __restrict__ dst, int n4) {
  int i = blockIdx.x * 256 + threadIdx.x;
  if (i < n4) {
    f32x4 s = *reinterpret_cast<const f32x4*>(src + (size_t)i * 4);
    bf16x4 d;
    d[0] = (bf16_t)s[0]; d[1] = (bf16_t)s[1];
    d[2] = (bf16_t)s[2]; d[3] = (bf16_t)s[3];
    *reinterpret_cast<bf16x4*>(dst + (size_t)i * 4) = d;
  }
}

// ------------------------- fused QKVS projection --------------------------
// out = x @ W^T : M=8192 (s*bsz), N=1024, K=1024. One wave -> 16x64 tile
// (4 N-tiles, A fragment reused x4), k-loop double-buffered.
__global__ __launch_bounds__(128) void proj_gemm_kernel(
    const bf16_t* __restrict__ xbf, const bf16_t* __restrict__ wbf,
    bf16_t* __restrict__ qbf, bf16_t* __restrict__ kbf,
    bf16_t* __restrict__ vbf, float* __restrict__ rbuf) {
  const int lane = threadIdx.x & 31;
  const int wave = threadIdx.x >> 5;
  const int kh = lane >> 4;          // K half of the wave
  const int ln = lane & 15;
  const int mtile = blockIdx.x;                   // 0..511
  const int nt0   = blockIdx.y * 16 + wave * 4;   // 4 adjacent 16-col tiles
  const int wsel  = blockIdx.z;                   // 0=q 1=k 2=v 3=r

  const bf16_t* W = wbf + (size_t)wsel * (EMB * EMB);
  // Hoisted pointers: inner loop uses pure immediate offsets.
  const bf16_t* ap = xbf + (size_t)(mtile * 16 + ln) * EMB + kh * 16;
  const bf16_t* bp0 = W + (size_t)((nt0 + 0) * 16 + ln) * EMB + kh * 16;
  const bf16_t* bp1 = W + (size_t)((nt0 + 1) * 16 + ln) * EMB + kh * 16;
  const bf16_t* bp2 = W + (size_t)((nt0 + 2) * 16 + ln) * EMB + kh * 16;
  const bf16_t* bp3 = W + (size_t)((nt0 + 3) * 16 + ln) * EMB + kh * 16;

  v8f acc[4] = {{}, {}, {}, {}};
  AB16 a[2];
  AB16 bfr[2][4];

  auto load_stage = [&](int buf, int k0) {
    a[buf].h[0] = *reinterpret_cast<const bf16x8*>(ap + k0);
    a[buf].h[1] = *reinterpret_cast<const bf16x8*>(ap + k0 + 8);
    bfr[buf][0].h[0] = *reinterpret_cast<const bf16x8*>(bp0 + k0);
    bfr[buf][0].h[1] = *reinterpret_cast<const bf16x8*>(bp0 + k0 + 8);
    bfr[buf][1].h[0] = *reinterpret_cast<const bf16x8*>(bp1 + k0);
    bfr[buf][1].h[1] = *reinterpret_cast<const bf16x8*>(bp1 + k0 + 8);
    bfr[buf][2].h[0] = *reinterpret_cast<const bf16x8*>(bp2 + k0);
    bfr[buf][2].h[1] = *reinterpret_cast<const bf16x8*>(bp2 + k0 + 8);
    bfr[buf][3].h[0] = *reinterpret_cast<const bf16x8*>(bp3 + k0);
    bfr[buf][3].h[1] = *reinterpret_cast<const bf16x8*>(bp3 + k0 + 8);
  };

  load_stage(0, 0);
#pragma unroll
  for (int kk = 0; kk < 32; ++kk) {           // k0 = kk*32
    const int cur = kk & 1, nxt = cur ^ 1;
    if (kk + 1 < 32) load_stage(nxt, (kk + 1) * 32);  // loads fly behind WMMAs
#pragma unroll
    for (int t = 0; t < 4; ++t)
      acc[t] = wmma_bf16(a[cur].v, bfr[cur][t].v, acc[t]);
  }

  for (int t = 0; t < 4; ++t) {
    const int nt = nt0 + t;
    for (int i = 0; i < 8; ++i) {
      const int m = i + kh * 8;
      const int grow = mtile * 16 + m;      // flattened (s,b)
      const int s = grow >> 4, bb = grow & 15;
      const int n = nt * 16 + ln;           // emb index
      const int h = n >> 6, d = n & 63;
      const float val = acc[t][i];
      if (wsel == 0) {        // q, pre-scaled by hd^-0.5 = 0.125
        qbf[(((size_t)bb * NHEAD + h) * SLEN + s) * HD + d] = (bf16_t)(val * 0.125f);
      } else if (wsel == 1) { // k
        kbf[(((size_t)bb * NHEAD + h) * SLEN + s) * HD + d] = (bf16_t)val;
      } else if (wsel == 2) { // v transposed: [b][h][d][s]
        vbf[(((size_t)bb * NHEAD + h) * HD + d) * SLEN + s] = (bf16_t)val;
      } else {                // r in natural (s,b,e) layout
        rbuf[((size_t)s * BSZ + bb) * EMB + n] = val;
      }
    }
  }
}

// --------------------- fused attention (flash style) ----------------------
// One wave per (b, h, 16-row q tile). Scores stripe 16x512 lives in LDS.
__global__ __launch_bounds__(32) void attn_fused_kernel(
    const bf16_t* __restrict__ qbf, const bf16_t* __restrict__ kbf,
    const bf16_t* __restrict__ vbf, const int* __restrict__ maskp,
    const int* __restrict__ wmp, float* __restrict__ obuf) {
  __shared__ float   scores[16][516];        // padded: bank spread, 16B-aligned rows
  __shared__ bf16_t  attnb[16][520];         // bf16 probs for WMMA A-frag
  __shared__ unsigned char mbits[16][512];   // bit0 = mask, bit1 = weight_mask

  const int lane = threadIdx.x & 31;
  const int kh = lane >> 4, ln = lane & 15;
  const int qt = blockIdx.x, h = blockIdx.y, b = blockIdx.z;

  const bf16_t* qbase = qbf + ((size_t)b * NHEAD + h) * SLEN * HD;
  const bf16_t* kbase = kbf + ((size_t)b * NHEAD + h) * SLEN * HD;
  const bf16_t* vbase = vbf + ((size_t)b * NHEAD + h) * HD * SLEN;
  const size_t mrowbase = ((size_t)b * SLEN + qt * 16) * SLEN;

  // Prefetch the two 32KB mask stripes so they are in-cache when the softmax
  // passes start (overlaps HBM latency with the score WMMAs below).
  {
    const char* m0 = (const char*)(maskp + mrowbase);
    const char* m1 = (const char*)(wmp + mrowbase);
#pragma unroll
    for (int i = 0; i < 4; ++i) {
      __builtin_prefetch(m0 + (size_t)(i * 32 + lane) * 256, 0, 1);
      __builtin_prefetch(m1 + (size_t)(i * 32 + lane) * 256, 0, 1);
    }
  }

  // ---- Stage 1: scores = (q*scale) @ k^T   (16 x 512), double-buffered ----
  {
    AB16 a0, a1;  // q fragments are loop-invariant over key tiles
    const bf16_t* qrowp = qbase + (size_t)(qt * 16 + ln) * HD + kh * 16;
    a0.h[0] = *reinterpret_cast<const bf16x8*>(qrowp);
    a0.h[1] = *reinterpret_cast<const bf16x8*>(qrowp + 8);
    a1.h[0] = *reinterpret_cast<const bf16x8*>(qrowp + 32);
    a1.h[1] = *reinterpret_cast<const bf16x8*>(qrowp + 32 + 8);

    const bf16_t* kp = kbase + (size_t)ln * HD + kh * 16;  // + kt*16*HD
    AB16 kb[2][2];
    auto load_k = [&](int buf, int kt) {
      const bf16_t* krowp = kp + (size_t)kt * (16 * HD);
      kb[buf][0].h[0] = *reinterpret_cast<const bf16x8*>(krowp);
      kb[buf][0].h[1] = *reinterpret_cast<const bf16x8*>(krowp + 8);
      kb[buf][1].h[0] = *reinterpret_cast<const bf16x8*>(krowp + 32);
      kb[buf][1].h[1] = *reinterpret_cast<const bf16x8*>(krowp + 32 + 8);
    };

    load_k(0, 0);
#pragma unroll
    for (int kt = 0; kt < 32; ++kt) {
      const int cur = kt & 1, nxt = cur ^ 1;
      if (kt + 1 < 32) load_k(nxt, kt + 1);
      v8f acc = {};
      acc = wmma_bf16(a0.v, kb[cur][0].v, acc);
      acc = wmma_bf16(a1.v, kb[cur][1].v, acc);
      for (int i = 0; i < 8; ++i)
        scores[i + kh * 8][kt * 16 + ln] = acc[i];
    }
  }
  __syncthreads();

  // ---- Stage 2: double masked softmax + degree normalization (4-wide) ----
  // Two lanes per row: lane ln -> cols [0,256), lane ln+16 -> cols [256,512).
  const int row = ln;
  const int cb = kh * 256;
  const size_t mrow = mrowbase + (size_t)row * SLEN;

  float mx1 = -3.0e38f;
  for (int t = 0; t < 64; ++t) {
    const int col = cb + t * 4;
    i32x4 mv4 = *reinterpret_cast<const i32x4*>(maskp + mrow + col);
    i32x4 wv4 = *reinterpret_cast<const i32x4*>(wmp + mrow + col);
    f32x4 sc4 = *reinterpret_cast<const f32x4*>(&scores[row][col]);
    unsigned int pk = 0;
#pragma unroll
    for (int j = 0; j < 4; ++j) {
      const int mv = mv4[j] != 0, wv = wv4[j] != 0;
      pk |= (unsigned int)(mv | (wv << 1)) << (8 * j);
      mx1 = fmaxf(mx1, mv ? sc4[j] : NEG1F);
    }
    *reinterpret_cast<unsigned int*>(&mbits[row][col]) = pk;
  }
  mx1 = fmaxf(mx1, __shfl_xor(mx1, 16));

  float sum1 = 0.f;
  for (int t = 0; t < 64; ++t) {
    const int col = cb + t * 4;
    const unsigned int pk = *reinterpret_cast<const unsigned int*>(&mbits[row][col]);
    f32x4 sc4 = *reinterpret_cast<const f32x4*>(&scores[row][col]);
    f32x4 e4;
#pragma unroll
    for (int j = 0; j < 4; ++j) {
      const unsigned int mb = (pk >> (8 * j)) & 3u;
      const float s1 = (mb & 1u) ? sc4[j] : NEG1F;
      const float e1 = __expf(s1 - mx1);
      sum1 += e1;
      e4[j] = (mb & 1u) ? e1 : 0.f;            // e1 * mask
    }
    *reinterpret_cast<f32x4*>(&scores[row][col]) = e4;
  }
  sum1 += __shfl_xor(sum1, 16);
  const float inv1 = 1.f / sum1;

  float mx2 = -3.0e38f;
  for (int t = 0; t < 64; ++t) {
    const int col = cb + t * 4;
    const unsigned int pk = *reinterpret_cast<const unsigned int*>(&mbits[row][col]);
    f32x4 e4 = *reinterpret_cast<const f32x4*>(&scores[row][col]);
#pragma unroll
    for (int j = 0; j < 4; ++j) {
      const unsigned int mb = (pk >> (8 * j)) & 3u;
      mx2 = fmaxf(mx2, (mb & 2u) ? e4[j] * inv1 : NEG2F);
    }
  }
  mx2 = fmaxf(mx2, __shfl_xor(mx2, 16));

  float sAll = 0.f, sWm = 0.f;
  for (int t = 0; t < 64; ++t) {
    const int col = cb + t * 4;
    const unsigned int pk = *reinterpret_cast<const unsigned int*>(&mbits[row][col]);
    f32x4 e4 = *reinterpret_cast<const f32x4*>(&scores[row][col]);
    f32x4 w4;
#pragma unroll
    for (int j = 0; j < 4; ++j) {
      const unsigned int mb = (pk >> (8 * j)) & 3u;
      const float s2 = (mb & 2u) ? e4[j] * inv1 : NEG2F;
      const float e2 = __expf(s2 - mx2);
      sAll += e2;
      const float ew = (mb & 2u) ? e2 : 0.f;
      sWm += ew;
      w4[j] = ew;
    }
    *reinterpret_cast<f32x4*>(&scores[row][col]) = w4;
  }
  sAll += __shfl_xor(sAll, 16);
  sWm  += __shfl_xor(sWm, 16);
  // final = (ew/sAll) / (sWm/sAll + 1e-10) = ew / (sWm + 1e-10*sAll)
  const float invd = 1.f / (sWm + 1e-10f * sAll);

  for (int t = 0; t < 64; ++t) {
    const int col = cb + t * 4;
    f32x4 w4 = *reinterpret_cast<const f32x4*>(&scores[row][col]);
    bf16x4 a4;
    a4[0] = (bf16_t)(w4[0] * invd); a4[1] = (bf16_t)(w4[1] * invd);
    a4[2] = (bf16_t)(w4[2] * invd); a4[3] = (bf16_t)(w4[3] * invd);
    *reinterpret_cast<bf16x4*>(&attnb[row][col]) = a4;
  }
  __syncthreads();

  // ---- Stage 3: out = attn @ v  (16 x 64), A from LDS, B = v^T global,
  //      v fragments double-buffered across the k loop ----
  for (int nt = 0; nt < 4; ++nt) {
    v8f acc = {};
    const bf16_t* vrowp = vbase + (size_t)(nt * 16 + ln) * SLEN + kh * 16;
    AB16 vb[2];
    auto load_v = [&](int buf, int kk) {
      vb[buf].h[0] = *reinterpret_cast<const bf16x8*>(vrowp + kk * 32);
      vb[buf].h[1] = *reinterpret_cast<const bf16x8*>(vrowp + kk * 32 + 8);
    };
    load_v(0, 0);
#pragma unroll
    for (int kk = 0; kk < 16; ++kk) {
      const int cur = kk & 1, nxt = cur ^ 1;
      if (kk + 1 < 16) load_v(nxt, kk + 1);
      AB16 a;
      const int ab = kk * 32 + kh * 16;
      a.h[0] = *reinterpret_cast<const bf16x8*>(&attnb[ln][ab]);
      a.h[1] = *reinterpret_cast<const bf16x8*>(&attnb[ln][ab + 8]);
      acc = wmma_bf16(a.v, vb[cur].v, acc);
    }
    for (int i = 0; i < 8; ++i) {
      const int s = qt * 16 + i + kh * 8;
      obuf[((size_t)s * BSZ + b) * EMB + h * HD + nt * 16 + ln] = acc[i];
    }
  }
}

// ------------------------------ gate combine ------------------------------
// gate = sigmoid([out, r, out-r] @ Wb^T)  (scalar per row); y = g*r+(1-g)*out
__global__ __launch_bounds__(128) void combine_kernel(
    const float* __restrict__ obuf, const float* __restrict__ rbuf,
    const float* __restrict__ wb, float* __restrict__ y) {
  const int row = blockIdx.x * 4 + (threadIdx.x >> 5);  // 0..8191 = s*16+b
  const int lane = threadIdx.x & 31;
  const float* orow = obuf + (size_t)row * EMB;
  const float* rrow = rbuf + (size_t)row * EMB;
  float dot = 0.f;
  for (int t = 0; t < 8; ++t) {
    const int e = t * 128 + lane * 4;
    f32x4 o  = *reinterpret_cast<const f32x4*>(orow + e);
    f32x4 rv = *reinterpret_cast<const f32x4*>(rrow + e);
    f32x4 w0 = *reinterpret_cast<const f32x4*>(wb + e);
    f32x4 w1 = *reinterpret_cast<const f32x4*>(wb + EMB + e);
    f32x4 w2 = *reinterpret_cast<const f32x4*>(wb + 2 * EMB + e);
#pragma unroll
    for (int j = 0; j < 4; ++j)
      dot += o[j] * w0[j] + rv[j] * w1[j] + (o[j] - rv[j]) * w2[j];
  }
  for (int off = 16; off; off >>= 1) dot += __shfl_xor(dot, off);
  const float g = 1.f / (1.f + __expf(-dot));
  for (int t = 0; t < 8; ++t) {
    const int e = t * 128 + lane * 4;
    f32x4 o  = *reinterpret_cast<const f32x4*>(orow + e);
    f32x4 rv = *reinterpret_cast<const f32x4*>(rrow + e);
    f32x4 res;
#pragma unroll
    for (int j = 0; j < 4; ++j)
      res[j] = g * rv[j] + (1.f - g) * o[j];
    *reinterpret_cast<f32x4*>(y + (size_t)row * EMB + e) = res;
  }
}

// ------------------------------ launcher ----------------------------------
extern "C" void kernel_launch(void* const* d_in, const int* in_sizes, int n_in,
                              void* d_out, int out_size, void* d_ws, size_t ws_size,
                              hipStream_t stream) {
  const float* x   = (const float*)d_in[0];
  const int* mask  = (const int*)d_in[1];
  const int* wmask = (const int*)d_in[2];
  const float* Wq  = (const float*)d_in[3];
  const float* Wk  = (const float*)d_in[4];
  const float* Wv  = (const float*)d_in[5];
  const float* Ws  = (const float*)d_in[6];
  const float* Wb  = (const float*)d_in[7];
  float* out = (float*)d_out;

  const size_t NX = (size_t)SLEN * BSZ * EMB;   // 8388608
  const size_t NW = (size_t)EMB * EMB;          // 1048576

  bf16_t* xbf = (bf16_t*)d_ws;
  bf16_t* wbf = xbf + NX;          // 4 weight matrices
  bf16_t* qbf = wbf + 4 * NW;
  bf16_t* kbf = qbf + NX;
  bf16_t* vbf = kbf + NX;
  float*  rbuf = (float*)(vbf + NX);
  float*  obuf = rbuf + NX;
  (void)ws_size; (void)in_sizes; (void)n_in; (void)out_size;

  // 1) fp32 -> bf16 copies of x and the 4 projection weights (vectorized x4)
  f32_to_bf16_kernel<<<(int)(NX / 1024), 256, 0, stream>>>(x, xbf, (int)(NX / 4));
  f32_to_bf16_kernel<<<(int)(NW / 1024), 256, 0, stream>>>(Wq, wbf + 0 * NW, (int)(NW / 4));
  f32_to_bf16_kernel<<<(int)(NW / 1024), 256, 0, stream>>>(Wk, wbf + 1 * NW, (int)(NW / 4));
  f32_to_bf16_kernel<<<(int)(NW / 1024), 256, 0, stream>>>(Wv, wbf + 2 * NW, (int)(NW / 4));
  f32_to_bf16_kernel<<<(int)(NW / 1024), 256, 0, stream>>>(Ws, wbf + 3 * NW, (int)(NW / 4));

  // 2) q/k/v/r projections (WMMA GEMM, 16x64 per wave, double-buffered)
  proj_gemm_kernel<<<dim3(512, 4, 4), 128, 0, stream>>>(xbf, wbf, qbf, kbf, vbf, rbuf);

  // 3) fused attention: scores + double masked softmax + degree norm + @v
  attn_fused_kernel<<<dim3(SLEN / 16, NHEAD, BSZ), 32, 0, stream>>>(
      qbf, kbf, vbf, mask, wmask, obuf);

  // 4) gated combine
  combine_kernel<<<dim3(SLEN * BSZ / 4), 128, 0, stream>>>(obuf, rbuf, Wb, out);
}
